// EdgeLogiter_89627377533191
// MI455X (gfx1250) — matrix operand order repaired and checked by reading, not compiled, not run on hardware
//
#include <hip/hip_runtime.h>
#include <hip/hip_bf16.h>

// Problem constants (from reference)
#define B_ 8
#define S_ 1024
#define H_ 256
#define C_ 4
#define THETA_ 10000.0f

typedef __attribute__((ext_vector_type(16))) __bf16 v16bf;
typedef __attribute__((ext_vector_type(8)))  float  v8f;

union V16 {
    v16bf v;
    uint4 q[2];
};

__device__ __forceinline__ unsigned short f2bf(float f) {
    unsigned int u = __float_as_uint(f);
    u += 0x7FFFu + ((u >> 16) & 1u);   // round-to-nearest-even
    return (unsigned short)(u >> 16);
}

// ---------------------------------------------------------------------------
// Stage 1: q = x@Wq + bq, k = x@Wk + bk, RoPE at +pos and -pos, store bf16.
// One block per (b,s) row. 256 threads: each thread computes 2 RoPE pairs
// (4 outputs) for q and for k.
// ---------------------------------------------------------------------------
__global__ void __launch_bounds__(256)
proj_rope_kernel(const float* __restrict__ x,
                 const float* __restrict__ Wq, const float* __restrict__ bq,
                 const float* __restrict__ Wk, const float* __restrict__ bk,
                 const int*   __restrict__ token_index,
                 unsigned short* __restrict__ qp, unsigned short* __restrict__ qn,
                 unsigned short* __restrict__ kp, unsigned short* __restrict__ kn)
{
    const int row = blockIdx.x;          // b*S + s
    const int t   = threadIdx.x;         // 0..255
    __shared__ float xs[H_];
    xs[t] = x[(size_t)row * H_ + t];
    __syncthreads();

    const int pos = token_index[row];
    const float2* __restrict__ wq2 = (const float2*)Wq;  // [256][512] pairs
    const float2* __restrict__ wk2 = (const float2*)Wk;

    #pragma unroll
    for (int half = 0; half < 2; ++half) {
        const int jb = half * 512 + 2 * t;     // even column in [0,1024)
        const int cp = jb >> 1;                // float2 column index

        float q0 = bq[jb], q1 = bq[jb + 1];
        float k0 = bk[jb], k1 = bk[jb + 1];
        #pragma unroll 8
        for (int k = 0; k < H_; ++k) {
            const float  xv = xs[k];
            const float2 wq = wq2[(size_t)k * 512 + cp];
            const float2 wk = wk2[(size_t)k * 512 + cp];
            q0 = __builtin_fmaf(xv, wq.x, q0);
            q1 = __builtin_fmaf(xv, wq.y, q1);
            k0 = __builtin_fmaf(xv, wk.x, k0);
            k1 = __builtin_fmaf(xv, wk.y, k1);
        }

        // RoPE: pair p of head c; freq = THETA^(-p/128)
        const int c = jb >> 8;        // jb / 256
        const int h = jb & 255;       // jb % 256 (even)
        const int p = h >> 1;
        const float freq = __powf(THETA_, -(float)p * (1.0f / 128.0f));
        const float ang  = (float)pos * freq;
        const float cs = __cosf(ang);
        const float sn = __sinf(ang);

        const size_t o = (((size_t)row * C_) + c) * H_ + h;   // even offset

        // +pos:  re = v0*cs - v1*sn ; im = v0*sn + v1*cs
        // -pos:  re = v0*cs + v1*sn ; im = -v0*sn + v1*cs
        unsigned int qpw = (unsigned int)f2bf(q0 * cs - q1 * sn) |
                           ((unsigned int)f2bf(q0 * sn + q1 * cs) << 16);
        unsigned int qnw = (unsigned int)f2bf(q0 * cs + q1 * sn) |
                           ((unsigned int)f2bf(q1 * cs - q0 * sn) << 16);
        unsigned int kpw = (unsigned int)f2bf(k0 * cs - k1 * sn) |
                           ((unsigned int)f2bf(k0 * sn + k1 * cs) << 16);
        unsigned int knw = (unsigned int)f2bf(k0 * cs + k1 * sn) |
                           ((unsigned int)f2bf(k1 * cs - k0 * sn) << 16);

        *(unsigned int*)(qp + o) = qpw;
        *(unsigned int*)(qn + o) = qnw;
        *(unsigned int*)(kp + o) = kpw;
        *(unsigned int*)(kn + o) = knw;
    }
}

// ---------------------------------------------------------------------------
// Stage 2: per (b,c): l1 = qp·kp^T, l2 = qn·kp^T, l3 = qp·kn^T via bf16 WMMA
// with fused thread_id select.
//
// Block = 256 threads = 8 waves arranged 2(m) x 4(n); block tile 64 x 128.
// LDS double buffer stages qp/qn (64x32) and kp/kn (128x32) per K-chunk
// (2 x 24 KB). Each wave computes a 32x32 macro-tile: 2x2 WMMA tiles x 3
// accumulator sets = 12 v_wmma per K-chunk, operands shared across tiles.
// ---------------------------------------------------------------------------

// LDS ushort-offsets within one buffer set (12288 ushorts = 24 KB)
#define LDS_AQ 0        // qp rows: 64 x 32
#define LDS_AN 2048     // qn rows: 64 x 32
#define LDS_BP 4096     // kp rows: 128 x 32
#define LDS_BN 8192     // kn rows: 128 x 32
#define LDS_SET 12288

__global__ void __launch_bounds__(256)
gemm_select_kernel(const unsigned short* __restrict__ qp,
                   const unsigned short* __restrict__ qn,
                   const unsigned short* __restrict__ kp,
                   const unsigned short* __restrict__ kn,
                   const int* __restrict__ thread_id,
                   float* __restrict__ out)
{
    const int bc   = blockIdx.z;          // b*C + c
    const int b    = bc >> 2;
    const int c    = bc & 3;
    const int tid  = threadIdx.x;
    const int lane = tid & 31;
    const int wave = tid >> 5;
    const int wm   = wave & 1;            // 2 waves along M
    const int wn   = wave >> 1;           // 4 waves along N
    const int mBlock = blockIdx.x * 64;
    const int nBlock = blockIdx.y * 128;
    const int l15  = lane & 15;
    const int hi   = lane >> 4;           // 0: lanes 0-15, 1: lanes 16-31

    __shared__ unsigned short lds[2 * LDS_SET];   // 48 KB double buffer

    // ---- staging map: 384 rows x 4 uint4 = 1536 uint4 per chunk; 6/thread
    const unsigned short* gptr[6];
    int lofs[6];
    #pragma unroll
    for (int j = 0; j < 6; ++j) {
        const int u  = tid + j * 256;
        const int r  = u >> 2;            // 0..383
        const int qd = u & 3;             // 16B quad within a 64B row
        const unsigned short* tsrc;
        int srow, loff;
        if (r < 64)       { tsrc = qp; srow = mBlock + r;         loff = LDS_AQ + r * 32; }
        else if (r < 128) { tsrc = qn; srow = mBlock + (r - 64);  loff = LDS_AN + (r - 64) * 32; }
        else if (r < 256) { tsrc = kp; srow = nBlock + (r - 128); loff = LDS_BP + (r - 128) * 32; }
        else              { tsrc = kn; srow = nBlock + (r - 256); loff = LDS_BN + (r - 256) * 32; }
        gptr[j] = tsrc + (((size_t)(b * S_ + srow)) * C_ + c) * H_ + qd * 8;
        lofs[j] = loff + qd * 8;
    }

    // ---- prime buffer 0 with K-chunk 0
    uint4 stg[6];
    #pragma unroll
    for (int j = 0; j < 6; ++j) stg[j] = *(const uint4*)(gptr[j]);
    #pragma unroll
    for (int j = 0; j < 6; ++j) *(uint4*)(&lds[lofs[j]]) = stg[j];
    __syncthreads();

    // ---- per-lane LDS read offsets (WMMA 16-bit A/B layouts, ISA 7.12.2)
    // A: lane row M=l15; lanes 0-15 K={0..7,16..23}, lanes 16-31 K={8..15,24..31}
    // B: lane col N=l15; lanes 0-15 K={0..7,16..23}, lanes 16-31 K={8..15,24..31}
    const int aoff = hi * 8;
    const int boff = hi * 8;
    const int arow0 = (wm * 32 + l15) * 32;
    const int arow1 = (wm * 32 + 16 + l15) * 32;
    const int brow0 = (wn * 32 + l15) * 32;
    const int brow1 = (wn * 32 + 16 + l15) * 32;

    v8f acc1[2][2] = {}, acc2[2][2] = {}, acc3[2][2] = {};

    for (int i = 0; i < 8; ++i) {                 // K chunks of 32
        const int cur = (i & 1) ? LDS_SET : 0;
        const int nxt = (i & 1) ? 0 : LDS_SET;

        if (i < 7) {                              // fetch chunk i+1 into regs
            #pragma unroll
            for (int j = 0; j < 6; ++j)
                stg[j] = *(const uint4*)(gptr[j] + (i + 1) * 32);
        }
        if (i < 6) {                              // pull chunk i+2 toward WGP
            __builtin_prefetch(gptr[0] + (i + 2) * 32, 0, 3);
            __builtin_prefetch(gptr[3] + (i + 2) * 32, 0, 3);
        }

        // ---- LDS -> VGPR tile reads
        V16 ap0, ap1, an0, an1, bp0, bp1, bn0, bn1;
        ap0.q[0] = *(const uint4*)(&lds[cur + LDS_AQ + arow0 + aoff]);
        ap0.q[1] = *(const uint4*)(&lds[cur + LDS_AQ + arow0 + aoff + 16]);
        ap1.q[0] = *(const uint4*)(&lds[cur + LDS_AQ + arow1 + aoff]);
        ap1.q[1] = *(const uint4*)(&lds[cur + LDS_AQ + arow1 + aoff + 16]);
        an0.q[0] = *(const uint4*)(&lds[cur + LDS_AN + arow0 + aoff]);
        an0.q[1] = *(const uint4*)(&lds[cur + LDS_AN + arow0 + aoff + 16]);
        an1.q[0] = *(const uint4*)(&lds[cur + LDS_AN + arow1 + aoff]);
        an1.q[1] = *(const uint4*)(&lds[cur + LDS_AN + arow1 + aoff + 16]);
        bp0.q[0] = *(const uint4*)(&lds[cur + LDS_BP + brow0 + boff]);
        bp0.q[1] = *(const uint4*)(&lds[cur + LDS_BP + brow0 + boff + 16]);
        bp1.q[0] = *(const uint4*)(&lds[cur + LDS_BP + brow1 + boff]);
        bp1.q[1] = *(const uint4*)(&lds[cur + LDS_BP + brow1 + boff + 16]);
        bn0.q[0] = *(const uint4*)(&lds[cur + LDS_BN + brow0 + boff]);
        bn0.q[1] = *(const uint4*)(&lds[cur + LDS_BN + brow0 + boff + 16]);
        bn1.q[0] = *(const uint4*)(&lds[cur + LDS_BN + brow1 + boff]);
        bn1.q[1] = *(const uint4*)(&lds[cur + LDS_BN + brow1 + boff + 16]);

        // ---- 12 WMMAs: 2x2 tiles x {l1,l2,l3}
        acc1[0][0] = __builtin_amdgcn_wmma_f32_16x16x32_bf16(false, ap0.v, false, bp0.v, (short)0, acc1[0][0], false, false);
        acc1[0][1] = __builtin_amdgcn_wmma_f32_16x16x32_bf16(false, ap0.v, false, bp1.v, (short)0, acc1[0][1], false, false);
        acc1[1][0] = __builtin_amdgcn_wmma_f32_16x16x32_bf16(false, ap1.v, false, bp0.v, (short)0, acc1[1][0], false, false);
        acc1[1][1] = __builtin_amdgcn_wmma_f32_16x16x32_bf16(false, ap1.v, false, bp1.v, (short)0, acc1[1][1], false, false);
        acc2[0][0] = __builtin_amdgcn_wmma_f32_16x16x32_bf16(false, an0.v, false, bp0.v, (short)0, acc2[0][0], false, false);
        acc2[0][1] = __builtin_amdgcn_wmma_f32_16x16x32_bf16(false, an0.v, false, bp1.v, (short)0, acc2[0][1], false, false);
        acc2[1][0] = __builtin_amdgcn_wmma_f32_16x16x32_bf16(false, an1.v, false, bp0.v, (short)0, acc2[1][0], false, false);
        acc2[1][1] = __builtin_amdgcn_wmma_f32_16x16x32_bf16(false, an1.v, false, bp1.v, (short)0, acc2[1][1], false, false);
        acc3[0][0] = __builtin_amdgcn_wmma_f32_16x16x32_bf16(false, ap0.v, false, bn0.v, (short)0, acc3[0][0], false, false);
        acc3[0][1] = __builtin_amdgcn_wmma_f32_16x16x32_bf16(false, ap0.v, false, bn1.v, (short)0, acc3[0][1], false, false);
        acc3[1][0] = __builtin_amdgcn_wmma_f32_16x16x32_bf16(false, ap1.v, false, bn0.v, (short)0, acc3[1][0], false, false);
        acc3[1][1] = __builtin_amdgcn_wmma_f32_16x16x32_bf16(false, ap1.v, false, bn1.v, (short)0, acc3[1][1], false, false);

        if (i < 7) {                              // store chunk i+1 into nxt
            #pragma unroll
            for (int j = 0; j < 6; ++j) *(uint4*)(&lds[nxt + lofs[j]]) = stg[j];
        }
        __syncthreads();
    }

    // ---- epilogue: fused thread_id select, fp32 store
    // C/D layout: lane col N=l15; VGPR g -> M = g (lanes 0-15) or g+8 (16-31)
    #pragma unroll
    for (int mt = 0; mt < 2; ++mt) {
        #pragma unroll
        for (int nt = 0; nt < 2; ++nt) {
            const int n  = nBlock + wn * 32 + nt * 16 + l15;
            const int tj = thread_id[b * S_ + n];
            #pragma unroll
            for (int g = 0; g < 8; ++g) {
                const int m  = mBlock + wm * 32 + mt * 16 + g + hi * 8;
                const int ti = thread_id[b * S_ + m];
                float v = acc1[mt][nt][g];
                if (ti > 0 && ti < tj)      v = acc2[mt][nt][g];
                else if (tj > 0 && ti > tj) v = acc3[mt][nt][g];
                out[(((size_t)bc) * S_ + m) * S_ + n] = v;
            }
        }
    }
}

extern "C" void kernel_launch(void* const* d_in, const int* in_sizes, int n_in,
                              void* d_out, int out_size, void* d_ws, size_t ws_size,
                              hipStream_t stream) {
    const float* x  = (const float*)d_in[0];
    const float* Wq = (const float*)d_in[1];
    const float* bq = (const float*)d_in[2];
    const float* Wk = (const float*)d_in[3];
    const float* bk = (const float*)d_in[4];
    const int* token_index = (const int*)d_in[5];
    const int* thread_id   = (const int*)d_in[6];
    float* out = (float*)d_out;

    const size_t N = (size_t)B_ * S_ * C_ * H_;   // 8M elements per tensor
    unsigned short* qp = (unsigned short*)d_ws;
    unsigned short* qn = qp + N;
    unsigned short* kp = qn + N;
    unsigned short* kn = kp + N;                  // total 64 MB of d_ws

    proj_rope_kernel<<<B_ * S_, 256, 0, stream>>>(
        x, Wq, bq, Wk, bk, token_index, qp, qn, kp, kn);

    dim3 grid(S_ / 64, S_ / 128, B_ * C_);
    gemm_select_kernel<<<grid, 256, 0, stream>>>(
        qp, qn, kp, kn, thread_id, out);
}
